// MoELayer_60370060312647
// MI455X (gfx1250) — matrix operand order repaired and checked by reading, not compiled, not run on hardware
//
#include <hip/hip_runtime.h>
#include <hip/hip_bf16.h>
#include <stdint.h>

typedef __attribute__((ext_vector_type(16))) __bf16 v16bf;
typedef __attribute__((ext_vector_type(8)))  __bf16 v8bf;
typedef __attribute__((ext_vector_type(4)))  __bf16 v4bf;
typedef __attribute__((ext_vector_type(8)))  float  v8f;

#define NUM_EXPERTS 8
#define IN_F 1024
#define OUT_F 1024
#define N_TOK 8192

#define BM 128
#define BN 128
#define KT 32
#define NSTAGE (NUM_EXPERTS * (IN_F / KT))   // 256 pipeline stages

// ---------------------------------------------------------------------------
// CDNA5 async copy: global -> LDS, 16B per lane, tracked with ASYNCcnt.
// GVS addressing: s[base64] + v[off32].  LDS dest offset in a VGPR (VDST).
// ---------------------------------------------------------------------------
__device__ __forceinline__ void async_copy_b128(unsigned lds_off,
                                                const void* sbase, int voff) {
    asm volatile("global_load_async_to_lds_b128 %0, %1, %2"
                 :: "v"(lds_off), "v"(voff), "s"(sbase) : "memory");
}
__device__ __forceinline__ void wait_async4() {
    asm volatile("s_wait_asynccnt 4" ::: "memory");
}
__device__ __forceinline__ void wait_async0() {
    asm volatile("s_wait_asynccnt 0" ::: "memory");
}

// ---------------------------------------------------------------------------
// fp32 -> bf16 (flat) for x
// ---------------------------------------------------------------------------
__global__ __launch_bounds__(256) void cvt_f32_bf16(const float* __restrict__ in,
                                                    __bf16* __restrict__ out,
                                                    int n4) {
    int i = blockIdx.x * blockDim.x + threadIdx.x;
    if (i < n4) {
        float4 v = ((const float4*)in)[i];
        v4bf o;
        o[0] = (__bf16)v.x; o[1] = (__bf16)v.y;
        o[2] = (__bf16)v.z; o[3] = (__bf16)v.w;
        *(v4bf*)(out + (size_t)i * 4) = o;
    }
}

// ---------------------------------------------------------------------------
// fp32 -> bf16 transposed weights:  wbt[e][n][k] = We[e][k][n]
// LDS-tiled 32x32 transpose so both global read and write are coalesced.
// ---------------------------------------------------------------------------
__global__ __launch_bounds__(256) void cvt_w_t(const float* __restrict__ We,
                                               __bf16* __restrict__ wbt) {
    __shared__ float t[32][33];
    const int e  = blockIdx.z;
    const int k0 = blockIdx.y * 32;
    const int n0 = blockIdx.x * 32;
    const int tx = threadIdx.x & 31;   // n on read, k on write
    const int ty = threadIdx.x >> 5;   // 8-row passes

    const float* src = We + ((size_t)e << 20);
#pragma unroll
    for (int j = 0; j < 32; j += 8)
        t[ty + j][tx] = src[(size_t)(k0 + ty + j) * OUT_F + n0 + tx];
    __syncthreads();

    __bf16* dst = wbt + ((size_t)e << 20);
#pragma unroll
    for (int j = 0; j < 32; j += 8) {
        int n = ty + j;
        dst[(size_t)(n0 + n) * IN_F + k0 + tx] = (__bf16)t[tx][n];
    }
}

// ---------------------------------------------------------------------------
// Gate: softmax(x @ Wg + bg) -> gate[N_TOK, 8]. One wave32 per token.
// ---------------------------------------------------------------------------
__global__ __launch_bounds__(256) void moe_gate(const float* __restrict__ x,
                                                const float* __restrict__ Wg,
                                                const float* __restrict__ bg,
                                                float* __restrict__ gate) {
    const int token = blockIdx.x * 8 + (threadIdx.x >> 5);
    const int lane  = threadIdx.x & 31;

    float acc[NUM_EXPERTS];
#pragma unroll
    for (int e = 0; e < NUM_EXPERTS; ++e) acc[e] = 0.0f;

    const float* xr = x + (size_t)token * IN_F;
    for (int k = lane; k < IN_F; k += 32) {
        float xv = xr[k];
        const float* wr = Wg + (size_t)k * NUM_EXPERTS;
#pragma unroll
        for (int e = 0; e < NUM_EXPERTS; ++e) acc[e] += xv * wr[e];
    }
#pragma unroll
    for (int e = 0; e < NUM_EXPERTS; ++e) {
#pragma unroll
        for (int off = 16; off > 0; off >>= 1)
            acc[e] += __shfl_xor(acc[e], off, 32);
    }
    float v[NUM_EXPERTS];
    float mx = -3.4e38f;
#pragma unroll
    for (int e = 0; e < NUM_EXPERTS; ++e) {
        v[e] = acc[e] + bg[e];
        mx = fmaxf(mx, v[e]);
    }
    float s = 0.0f;
#pragma unroll
    for (int e = 0; e < NUM_EXPERTS; ++e) { v[e] = __expf(v[e] - mx); s += v[e]; }
    float inv = 1.0f / s;
    if (lane == 0) {
#pragma unroll
        for (int e = 0; e < NUM_EXPERTS; ++e)
            gate[(size_t)token * NUM_EXPERTS + e] = v[e] * inv;
    }
}

// ---------------------------------------------------------------------------
// Fused MoE GEMM + gated combine, async double-buffered.
//   out[n,:] = sum_e gate[n,e] * (x[n,:] @ We[e] + be[e])
// WG: 256 threads = 8 wave32, tile 128(M) x 128(N); waves 2(M) x 4(N),
// each wave 64x32 = 4x2 WMMA fragments -> 8 v_wmma per stage per wave.
// 256 stages = 8 experts x 32 k-steps; stage s+2 prefetched via
// global_load_async_to_lds_b128 while stage s computes.
// ---------------------------------------------------------------------------
__global__ __launch_bounds__(256) void moe_gemm(const __bf16* __restrict__ xb,
                                                const __bf16* __restrict__ wbt,
                                                const float*  __restrict__ be,
                                                const float*  __restrict__ gate,
                                                float* __restrict__ out) {
    __shared__ __bf16 lA[2][BM * KT];   // A tiles, row-major [m][k]  (2 x 8 KB)
    __shared__ __bf16 lB[2][BN * KT];   // B tiles, row-major [n][k]  (2 x 8 KB)
    __shared__ float  lG[BM * 8];       // gate tile [row][expert]    (4 KB)

    const int tid   = threadIdx.x;
    const int lane  = tid & 31;
    const int wave  = tid >> 5;
    const int waveM = wave & 1;        // 0..1 -> 64-row halves
    const int waveN = wave >> 1;       // 0..3 -> 32-col quarters
    const int hi    = lane >> 4;
    const int l16   = lane & 15;

    const int blockCol = blockIdx.x * BN;
    const int blockRow = blockIdx.y * BM;

    // ---- per-thread async-copy constants (4 x 16B chunks per stage) ----
    // chunks c0=tid, c1=tid+256 cover the 512 16B chunks of an 8 KB tile;
    // same decomposition reused for A (row m) and B (row n).
    const int c0 = tid, c1 = tid + 256;
    const int m0 = c0 >> 2, q0 = c0 & 3;
    const int m1 = c1 >> 2, q1 = c1 & 3;

    const unsigned lA0 = (unsigned)(uintptr_t)&lA[0][0] + (unsigned)((m0 * KT + q0 * 8) * 2);
    const unsigned lA1 = (unsigned)(uintptr_t)&lA[0][0] + (unsigned)((m1 * KT + q1 * 8) * 2);
    const unsigned lB0 = (unsigned)(uintptr_t)&lB[0][0] + (unsigned)((m0 * KT + q0 * 8) * 2);
    const unsigned lB1 = (unsigned)(uintptr_t)&lB[0][0] + (unsigned)((m1 * KT + q1 * 8) * 2);
    const int voffA0 = (m0 * IN_F + q0 * 8) * 2;    // bytes
    const int voffA1 = (m1 * IN_F + q1 * 8) * 2;
    const int voffB0 = voffA0;                       // wbt rows also IN_F-strided
    const int voffB1 = voffA1;

    const void* sA = (const void*)(xb  + (size_t)blockRow * IN_F);
    const void* sB = (const void*)(wbt + (size_t)blockCol * IN_F);

    const unsigned BUFB = (unsigned)(BM * KT * 2);   // 8192-byte buffer stride

    auto issue = [&](int s) {
        const unsigned bo = (unsigned)(s & 1) * BUFB;
        const int kb = (s & 31) * (KT * 2);               // k0 bytes
        const int eb = (s >> 5) * (IN_F * OUT_F * 2);     // expert bytes (B only)
        async_copy_b128(lA0 + bo, sA, voffA0 + kb);
        async_copy_b128(lA1 + bo, sA, voffA1 + kb);
        async_copy_b128(lB0 + bo, sB, voffB0 + kb + eb);
        async_copy_b128(lB1 + bo, sB, voffB1 + kb + eb);
    };

    // prologue: stages 0 and 1 in flight before anything else
    issue(0);
    issue(1);

    // stage gate tile (flat copy; first consumer is stage 31, many barriers away)
    for (int i = tid; i < BM * 8; i += 256)
        lG[i] = gate[(size_t)blockRow * 8 + i];

    v8f accT[4][2], accE[4][2];
#pragma unroll
    for (int mi = 0; mi < 4; ++mi)
#pragma unroll
        for (int ni = 0; ni < 2; ++ni)
#pragma unroll
            for (int r = 0; r < 8; ++r) { accT[mi][ni][r] = 0.0f; accE[mi][ni][r] = 0.0f; }

    for (int s = 0; s < NSTAGE; ++s) {
        // retire stage-s copies for this wave, then make them WG-visible
        if (s + 1 < NSTAGE) wait_async4(); else wait_async0();
        __syncthreads();

        const __bf16* Ab = &lA[s & 1][0];
        const __bf16* Bb = &lB[s & 1][0];

        v16bf aF[4];
#pragma unroll
        for (int mi = 0; mi < 4; ++mi) {
            // A per-lane: row = l16; K contiguous hi*16..hi*16+15 (matches B)
            const __bf16* ar = &Ab[(waveM * 64 + mi * 16 + l16) * KT];
            aF[mi] = *(const v16bf*)(ar + hi * 16);
        }
        v16bf bF[2];
#pragma unroll
        for (int ni = 0; ni < 2; ++ni)
            bF[ni] = *(const v16bf*)(&Bb[(waveN * 32 + ni * 16 + l16) * KT + hi * 16]);

#pragma unroll
        for (int mi = 0; mi < 4; ++mi)
#pragma unroll
            for (int ni = 0; ni < 2; ++ni)
                accE[mi][ni] = __builtin_amdgcn_wmma_f32_16x16x32_bf16(
                    false, aF[mi], false, bF[ni],
                    (short)0, accE[mi][ni], false, false);

        // expert boundary: accT += g * (accE + bias); reset accE
        if ((s & 31) == 31) {
            const int e = s >> 5;
            float bv[2];
#pragma unroll
            for (int ni = 0; ni < 2; ++ni)
                bv[ni] = be[(size_t)e * OUT_F + blockCol + waveN * 32 + ni * 16 + l16];
#pragma unroll
            for (int mi = 0; mi < 4; ++mi) {
#pragma unroll
                for (int r = 0; r < 8; ++r) {
                    int lr  = waveM * 64 + mi * 16 + r + 8 * hi;  // C/D row = r + 8*hi
                    float g = lG[lr * 8 + e];
#pragma unroll
                    for (int ni = 0; ni < 2; ++ni) {
                        accT[mi][ni][r] += g * (accE[mi][ni][r] + bv[ni]);
                        accE[mi][ni][r] = 0.0f;
                    }
                }
            }
        }

        // everyone done reading buf (s&1); safe to overwrite it with stage s+2
        __syncthreads();
        if (s + 2 < NSTAGE) issue(s + 2);
    }

    // ---- store ----
#pragma unroll
    for (int mi = 0; mi < 4; ++mi)
#pragma unroll
        for (int ni = 0; ni < 2; ++ni)
#pragma unroll
            for (int r = 0; r < 8; ++r) {
                int row = blockRow + waveM * 64 + mi * 16 + r + 8 * hi;
                int col = blockCol + waveN * 32 + ni * 16 + l16;
                out[(size_t)row * OUT_F + col] = accT[mi][ni][r];
            }
}

// ---------------------------------------------------------------------------
extern "C" void kernel_launch(void* const* d_in, const int* in_sizes, int n_in,
                              void* d_out, int out_size, void* d_ws, size_t ws_size,
                              hipStream_t stream) {
    const float* x  = (const float*)d_in[0];
    const float* We = (const float*)d_in[1];
    const float* be = (const float*)d_in[2];
    const float* Wg = (const float*)d_in[3];
    const float* bg = (const float*)d_in[4];
    float* out = (float*)d_out;

    char* ws = (char*)d_ws;
    float*  gate = (float*)ws;                                          // 256 KB
    __bf16* xb   = (__bf16*)(ws + (size_t)N_TOK * NUM_EXPERTS * sizeof(float));
    __bf16* wbt  = xb + (size_t)N_TOK * IN_F;                           // +16 MB

    int n4x = N_TOK * IN_F / 4;
    cvt_f32_bf16<<<(n4x + 255) / 256, 256, 0, stream>>>(x, xb, n4x);

    dim3 tgrid(OUT_F / 32, IN_F / 32, NUM_EXPERTS);
    cvt_w_t<<<tgrid, 256, 0, stream>>>(We, wbt);

    moe_gate<<<N_TOK / 8, 256, 0, stream>>>(x, Wg, bg, gate);

    dim3 grid(OUT_F / BN, N_TOK / BM);
    moe_gemm<<<grid, 256, 0, stream>>>(xb, wbt, be, gate, out);
}
